// TransformerXL_53206054863365
// MI455X (gfx1250) — compile-verified, matrix operand order checked by reading, not run on hardware
//
#include <hip/hip_runtime.h>
#include <hip/hip_bf16.h>

typedef __bf16 bf16;
typedef __attribute__((ext_vector_type(16))) __bf16 v16bf;
typedef __attribute__((ext_vector_type(8)))  float  v8f;

namespace {

constexpr int L_   = 8;
constexpr int DM   = 1024;
constexpr int NHD  = 16;
constexpr int DH   = 64;
constexpr int DI   = 4096;
constexpr int QL   = 512;
constexpr int ML   = 512;
constexpr int KL   = 1024;   // ML + QL
constexpr int BS   = 16;
constexpr int HID  = 1024;
constexpr int H3   = 3072;
constexpr int TH_N = 1000;
constexpr int TH_D = 256;
constexpr int RO_N = 64;
constexpr int RO_D = 128;
constexpr int FO_N = 32;
constexpr int FO_D = 128;
constexpr float LN_EPS = 1e-3f;

union Frag16 { uint4 u[2]; v16bf v; };

// ---------------------------------------------------------------------------
// bf16 WMMA GEMM with pre-transposed B:  C[z] = act(A[z] (MxK) @ Bt[z]^T + bias)
//   A  : [M][K] row-major bf16 (lda = K-stride)
//   Bt : [N][K] row-major bf16 (ldb = K-stride)  -> C[m][n] = sum_k A[m][k]*Bt[n][k]
// Tiles BM x BN x 64; 256 threads = 8 wave32 waves (4 m x 2 n).
// Double-buffered LDS; tiles staged with GLOBAL_LOAD_ASYNC_TO_LDS_B128
// (ASYNCcnt) so the async engine fills buffer[1-cur] while WMMA consumes
// buffer[cur]: one s_wait_asynccnt + one barrier per 64-wide k step.
// Out-of-range rows are CLAMPED (not zeroed): duplicated rows only affect
// C rows/cols >= M/N which the guarded epilogue never stores.
// ---------------------------------------------------------------------------
template<int BM, int BN, int OUT_BF16, int ACT>
__global__ __launch_bounds__(256) void gemm_k(
    const bf16* __restrict__ A, const bf16* __restrict__ Bt,
    void* __restrict__ Cv, const float* __restrict__ bias,
    int M, int N, int K, int lda, int ldb, int ldc,
    long sA, long sB, long sC)
{
  constexpr int BK  = 64;            // two WMMA k-steps per staging phase
  constexpr int PAD = 8;             // rows stay 16B-aligned (72*2 = 144 B)
  constexpr int MT  = BM / 64;       // 16-row tiles per wave (4 waves in m)
  constexpr int NT  = BN / 32;       // 16-col tiles per wave (2 waves in n)
  __shared__ bf16 As[2][BM][BK + PAD];
  __shared__ bf16 Bs[2][BN][BK + PAD];

  const int z = blockIdx.z;
  A  += z * sA;
  Bt += z * sB;
  float* Cf = (float*)Cv + z * sC;
  bf16*  Cb = (bf16*)Cv + z * sC;

  const int bm   = blockIdx.y * BM;
  const int bn   = blockIdx.x * BN;
  const int tid  = threadIdx.x;
  const int lane = tid & 31;
  const int wave = tid >> 5;
  const int l16  = lane & 15;
  const int lhi  = lane >> 4;
  const int wm   = wave & 3;
  const int wn   = wave >> 2;

  v8f acc[MT][NT];
#pragma unroll
  for (int mt = 0; mt < MT; ++mt)
#pragma unroll
    for (int nt = 0; nt < NT; ++nt)
#pragma unroll
      for (int e = 0; e < 8; ++e) acc[mt][nt][e] = 0.0f;

  // async stage one BK-wide tile pair into LDS buffer `buf`
  auto stage = [&](int k0, int buf) {
#pragma unroll
    for (int idx = tid; idx < BM * (BK / 8); idx += 256) {
      int r = idx >> 3, c8 = idx & 7;
      int gm = bm + r; gm = (gm < M) ? gm : (M - 1);
      const bf16* gp = A + (long)gm * lda + (k0 + c8 * 8);
      unsigned lo = (unsigned)(size_t)&As[buf][r][c8 * 8];
      asm volatile("global_load_async_to_lds_b128 %0, %1, off"
                   :: "v"(lo), "v"(gp) : "memory");
    }
#pragma unroll
    for (int idx = tid; idx < BN * (BK / 8); idx += 256) {
      int r = idx >> 3, c8 = idx & 7;
      int gn = bn + r; gn = (gn < N) ? gn : (N - 1);
      const bf16* gp = Bt + (long)gn * ldb + (k0 + c8 * 8);
      unsigned lo = (unsigned)(size_t)&Bs[buf][r][c8 * 8];
      asm volatile("global_load_async_to_lds_b128 %0, %1, off"
                   :: "v"(lo), "v"(gp) : "memory");
    }
  };

  stage(0, 0);
  int cur = 0;
  for (int k0 = 0; k0 < K; k0 += BK) {
    asm volatile("s_wait_asynccnt 0" ::: "memory");
    __syncthreads();                       // buffer[cur] visible to all waves
    if (k0 + BK < K) stage(k0 + BK, cur ^ 1);   // overlaps with math below

#pragma unroll
    for (int ks = 0; ks < BK / 32; ++ks) {
      v16bf af[MT];
      v16bf bfv[NT];
#pragma unroll
      for (int mt = 0; mt < MT; ++mt) {
        const int row = wm * (MT * 16) + mt * 16 + l16;
        const bf16* p = &As[cur][row][ks * 32 + lhi * 8];
        Frag16 f;
        f.u[0] = *(const uint4*)(p);
        f.u[1] = *(const uint4*)(p + 16);
        af[mt] = f.v;
      }
#pragma unroll
      for (int nt = 0; nt < NT; ++nt) {
        const int col = wn * (NT * 16) + nt * 16 + l16;
        const bf16* p = &Bs[cur][col][ks * 32 + lhi * 8];
        Frag16 f;
        f.u[0] = *(const uint4*)(p);
        f.u[1] = *(const uint4*)(p + 16);
        bfv[nt] = f.v;
      }
#pragma unroll
      for (int mt = 0; mt < MT; ++mt)
#pragma unroll
        for (int nt = 0; nt < NT; ++nt)
          acc[mt][nt] = __builtin_amdgcn_wmma_f32_16x16x32_bf16(
              false, af[mt], false, bfv[nt], (short)0, acc[mt][nt], false, false);
    }
    cur ^= 1;
  }

#pragma unroll
  for (int mt = 0; mt < MT; ++mt)
#pragma unroll
    for (int nt = 0; nt < NT; ++nt)
#pragma unroll
      for (int v = 0; v < 8; ++v) {
        int m = bm + wm * (MT * 16) + mt * 16 + lhi * 8 + v;
        int n = bn + wn * (NT * 16) + nt * 16 + l16;
        if (m < M && n < N) {
          float val = acc[mt][nt][v];
          if (bias) val += bias[n];
          if (ACT == 1) val = fmaxf(val, 0.0f);
          if (ACT == 2) val = tanhf(val);
          if (OUT_BF16) Cb[(long)m * ldc + n] = (bf16)val;
          else          Cf[(long)m * ldc + n] = val;
        }
      }
}

// ---------------------------------------------------------------------------
// Transpose + fp32->bf16 convert: dst[n][k] = (bf16)src[k][n], per-z slab.
// ---------------------------------------------------------------------------
__global__ __launch_bounds__(256) void k_f2bT(
    const float* __restrict__ src, bf16* __restrict__ dst,
    int K, int N, long sSrc, long sDst)
{
  __shared__ float tile[32][33];
  const float* s = src + (long)blockIdx.z * sSrc;
  bf16* d = dst + (long)blockIdx.z * sDst;
  const int k0 = blockIdx.y * 32, n0 = blockIdx.x * 32;
  const int tx = threadIdx.x & 31, ty = threadIdx.x >> 5;
#pragma unroll
  for (int yy = ty; yy < 32; yy += 8) {
    int k = k0 + yy, n = n0 + tx;
    tile[yy][tx] = (k < K && n < N) ? s[(long)k * N + n] : 0.0f;
  }
  __syncthreads();
#pragma unroll
  for (int yy = ty; yy < 32; yy += 8) {
    int n = n0 + yy, k = k0 + tx;
    if (n < N && k < K) d[(long)n * K + k] = (bf16)tile[tx][yy];
  }
}

// ---------------------------------------------------------------------------
// Elementwise / helper kernels
// ---------------------------------------------------------------------------
__global__ void k_f2b(const float* __restrict__ x, bf16* __restrict__ y, long n) {
  long i = (long)blockIdx.x * 256 + threadIdx.x;
  long stride = (long)gridDim.x * 256;
  for (; i < n; i += stride) y[i] = (bf16)x[i];
}

__global__ void k_embed(const int* __restrict__ dec, const float* __restrict__ wemb,
                        float* __restrict__ out) {
  long i = (long)blockIdx.x * 256 + threadIdx.x;     // QL*BS*DM
  long row = i / DM; int d = (int)(i % DM);
  int tok = dec[row];
  out[i] = wemb[(long)tok * DM + d];
}

__global__ void k_pos(bf16* __restrict__ posb) {
  int i = blockIdx.x * 256 + threadIdx.x;            // KL*512
  int j = i >> 9; int c = i & 511;
  float p = (float)(KL - 1 - j);
  float inv_freq = __expf(-((float)(2 * c) / (float)DM) * logf(10000.0f));
  float s = p * inv_freq;
  posb[(long)j * DM + c]       = (bf16)sinf(s);
  posb[(long)j * DM + 512 + c] = (bf16)cosf(s);
}

__global__ void k_cat(const float* __restrict__ mem_i, const float* __restrict__ outp,
                      bf16* __restrict__ catb) {
  long i = (long)blockIdx.x * 256 + threadIdx.x;     // KL*BS*DM
  long row = i / DM;
  int j = (int)(row / BS);
  float v = (j < ML) ? mem_i[i] : outp[i - (long)ML * BS * DM];
  catb[i] = (bf16)v;
}

// r_bf[h][j][d] = rh[j][h*64+d]  (coalesced both sides; natural Bt layout)
__global__ void k_prep_r(const float* __restrict__ rh, bf16* __restrict__ rb) {
  long i = (long)blockIdx.x * 256 + threadIdx.x;     // NHD*KL*DH
  int d = (int)(i % DH);
  int j = (int)((i / DH) % KL);
  int n = (int)(i / ((long)DH * KL));
  rb[i] = (bf16)rh[(long)j * DM + n * DH + d];
}

// q (+biases) and K extraction; K stays in natural [b][h][j][d] (Bt layout).
__global__ void k_prep_qk(const float* __restrict__ heads,
                          const float* __restrict__ rwb, const float* __restrict__ rrb,
                          bf16* __restrict__ qw, bf16* __restrict__ qr,
                          bf16* __restrict__ kb) {
  long i = (long)blockIdx.x * 256 + threadIdx.x;     // BS*NHD*KL*DH
  int d = (int)(i % DH);
  int j = (int)((i / DH) % KL);
  int n = (int)((i / ((long)DH * KL)) % NHD);
  int b = (int)(i / ((long)DH * KL * NHD));
  long hrow = ((long)j * BS + b) * H3;
  kb[i] = (bf16)heads[hrow + HID + n * DH + d];       // kb[b][n][j][d]
  if (j < QL) {
    float qv = heads[((long)(ML + j) * BS + b) * H3 + n * DH + d];
    long qidx = (((long)b * NHD + n) * QL + j) * DH + d;
    qw[qidx] = (bf16)(qv + rwb[n * DH + d]);
    qr[qidx] = (bf16)(qv + rrb[n * DH + d]);
  }
}

// vT[b][h][d][j] = heads.v[b][h][j][d]  via 32x32 LDS tile transpose
__global__ __launch_bounds__(256) void k_trans_v(const float* __restrict__ heads,
                                                 bf16* __restrict__ vT) {
  __shared__ float t[32][33];
  const int z = blockIdx.z;            // b*NHD + n
  const int b = z >> 4, n = z & 15;
  const int j0 = blockIdx.x * 32, d0 = blockIdx.y * 32;
  const int tx = threadIdx.x & 31, ty = threadIdx.x >> 5;
#pragma unroll
  for (int yy = ty; yy < 32; yy += 8) {
    int j = j0 + yy, d = d0 + tx;
    t[yy][tx] = heads[((long)j * BS + b) * H3 + 2 * HID + n * DH + d];
  }
  __syncthreads();
#pragma unroll
  for (int yy = ty; yy < 32; yy += 8) {
    int d = d0 + yy, j = j0 + tx;
    vT[(((long)b * NHD + n) * DH + d) * KL + j] = (bf16)t[tx][yy];
  }
}

// Exact Transformer-XL rel_shift + key mask + scale + softmax, probs -> bf16.
__global__ __launch_bounds__(256) void k_softmax(
    const float* __restrict__ S1, const float* __restrict__ S2,
    const int* __restrict__ dec, int b, bf16* __restrict__ P)
{
  const int n  = blockIdx.x >> 9;
  const int qi = blockIdx.x & (QL - 1);
  const int tid = threadIdx.x;
  const float scale = 0.125f;                        // 1/sqrt(64)
  __shared__ float red[256];
  const float* ac = S1 + ((long)n * QL + qi) * KL;

  float vals[4];
  float lmax = -3.0e38f;
#pragma unroll
  for (int t = 0; t < 4; ++t) {
    int j = tid + t * 256;
    int tt = qi * KL + j;                            // exact flat-index rel_shift
    int r = tt / QL, c = tt % QL;
    int s = (r + 1) * QL + c;
    int i2 = s / (KL + 1), j2 = s % (KL + 1);
    float bd = (j2 == 0) ? 0.0f : S2[((long)n * QL + i2) * KL + (j2 - 1)];
    bool keep = (j < ML) || (dec[(long)(j - ML) * BS + b] != 0);
    float sc = keep ? (ac[j] + bd) * scale : -1.0e9f;
    vals[t] = sc;
    lmax = fmaxf(lmax, sc);
  }
  red[tid] = lmax; __syncthreads();
  for (int s = 128; s > 0; s >>= 1) { if (tid < s) red[tid] = fmaxf(red[tid], red[tid + s]); __syncthreads(); }
  float mx = red[0]; __syncthreads();

  float lsum = 0.0f;
#pragma unroll
  for (int t = 0; t < 4; ++t) { vals[t] = __expf(vals[t] - mx); lsum += vals[t]; }
  red[tid] = lsum; __syncthreads();
  for (int s = 128; s > 0; s >>= 1) { if (tid < s) red[tid] += red[tid + s]; __syncthreads(); }
  float inv = 1.0f / red[0];

  bf16* prow = P + ((long)n * QL + qi) * KL;
#pragma unroll
  for (int t = 0; t < 4; ++t) prow[tid + t * 256] = (bf16)(vals[t] * inv);
}

// h = LN(x+y; g1,b1) -> f32;  abuf = LN(h; g2,b2) -> bf16 (one row per block)
__global__ __launch_bounds__(256) void k_add_ln2(
    const float* __restrict__ x, const float* __restrict__ y,
    const float* __restrict__ g1, const float* __restrict__ b1,
    const float* __restrict__ g2, const float* __restrict__ b2,
    float* __restrict__ h, bf16* __restrict__ abuf)
{
  const long row = blockIdx.x;
  const int tid = threadIdx.x;
  __shared__ float red[256];
  auto bsum = [&](float v) -> float {
    red[tid] = v; __syncthreads();
    for (int s = 128; s > 0; s >>= 1) { if (tid < s) red[tid] += red[tid + s]; __syncthreads(); }
    float r = red[0]; __syncthreads();
    return r;
  };
  const float* xr = x + row * DM;
  const float* yr = y + row * DM;
  float v[4]; float s = 0.0f;
#pragma unroll
  for (int t = 0; t < 4; ++t) { v[t] = xr[tid + t * 256] + yr[tid + t * 256]; s += v[t]; }
  float mu = bsum(s) * (1.0f / DM);
  float sv = 0.0f;
#pragma unroll
  for (int t = 0; t < 4; ++t) { float d = v[t] - mu; sv += d * d; }
  float rstd = rsqrtf(bsum(sv) * (1.0f / DM) + LN_EPS);
  float hv[4];
#pragma unroll
  for (int t = 0; t < 4; ++t) {
    int c = tid + t * 256;
    hv[t] = (v[t] - mu) * rstd * g1[c] + b1[c];
    h[row * DM + c] = hv[t];
  }
  s = 0.0f;
#pragma unroll
  for (int t = 0; t < 4; ++t) s += hv[t];
  mu = bsum(s) * (1.0f / DM);
  sv = 0.0f;
#pragma unroll
  for (int t = 0; t < 4; ++t) { float d = hv[t] - mu; sv += d * d; }
  rstd = rsqrtf(bsum(sv) * (1.0f / DM) + LN_EPS);
#pragma unroll
  for (int t = 0; t < 4; ++t) {
    int c = tid + t * 256;
    abuf[row * DM + c] = (bf16)((hv[t] - mu) * rstd * g2[c] + b2[c]);
  }
}

// out = LN(x+y; g,b) (f32)
__global__ __launch_bounds__(256) void k_add_ln1(
    const float* __restrict__ x, const float* __restrict__ y,
    const float* __restrict__ g, const float* __restrict__ b,
    float* __restrict__ outp)
{
  const long row = blockIdx.x;
  const int tid = threadIdx.x;
  __shared__ float red[256];
  auto bsum = [&](float v) -> float {
    red[tid] = v; __syncthreads();
    for (int s = 128; s > 0; s >>= 1) { if (tid < s) red[tid] += red[tid + s]; __syncthreads(); }
    float r = red[0]; __syncthreads();
    return r;
  };
  const float* xr = x + row * DM;
  const float* yr = y + row * DM;
  float v[4]; float s = 0.0f;
#pragma unroll
  for (int t = 0; t < 4; ++t) { v[t] = xr[tid + t * 256] + yr[tid + t * 256]; s += v[t]; }
  float mu = bsum(s) * (1.0f / DM);
  float sv = 0.0f;
#pragma unroll
  for (int t = 0; t < 4; ++t) { float d = v[t] - mu; sv += d * d; }
  float rstd = rsqrtf(bsum(sv) * (1.0f / DM) + LN_EPS);
#pragma unroll
  for (int t = 0; t < 4; ++t) {
    int c = tid + t * 256;
    outp[row * DM + c] = (v[t] - mu) * rstd * g[c] + b[c];
  }
}

// role_cat[64][256] = [role_emb | rel_r_f @ form_emb]
__global__ void k_role(const float* __restrict__ rel_r_f, const float* __restrict__ form_emb,
                       const float* __restrict__ role_emb, float* __restrict__ role_cat) {
  int i = blockIdx.x * 256 + threadIdx.x;           // 64*256
  int r = i >> 8; int c = i & 255;
  if (c < RO_D) { role_cat[i] = role_emb[r * RO_D + c]; return; }
  int f = c - RO_D;
  float s = 0.0f;
  for (int k = 0; k < FO_N; ++k) s += rel_r_f[r * FO_N + k] * form_emb[k * FO_D + f];
  role_cat[i] = s;
}

// theme_cat[1000][512] = [theme_emb | rel_t_r @ role_cat]  (bf16 GEMM A-side)
__global__ void k_theme_cat(const float* __restrict__ rel_t_r, const float* __restrict__ role_cat,
                            const float* __restrict__ theme_emb, bf16* __restrict__ tc) {
  long i = (long)blockIdx.x * 256 + threadIdx.x;    // 1000*512
  if (i >= (long)TH_N * 512) return;
  int t = (int)(i >> 9); int c = (int)(i & 511);
  float v;
  if (c < TH_D) v = theme_emb[(long)t * TH_D + c];
  else {
    int cc = c - TH_D;
    float s = 0.0f;
    for (int k = 0; k < RO_N; ++k) s += rel_t_r[(long)t * RO_N + k] * role_cat[k * 256 + cc];
    v = s;
  }
  tc[i] = (bf16)v;
}

__global__ void k_pool(const int* __restrict__ dec, const float* __restrict__ outp,
                       float* __restrict__ pooled) {
  int i = blockIdx.x * 256 + threadIdx.x;           // BS*DM
  int b = i / DM; int d = i % DM;
  float s = 0.0f;
  for (int q = 0; q < QL; ++q)
    if (dec[q * BS + b] == 2) s += outp[((long)q * BS + b) * DM + d];
  pooled[i] = s;
}

__global__ void k_logits(const float* __restrict__ pooled, const float* __restrict__ theme,
                         float* __restrict__ outp) {
  long i = (long)blockIdx.x * 256 + threadIdx.x;    // BS*TH_N
  if (i >= (long)BS * TH_N) return;
  int b = (int)(i / TH_N); int t = (int)(i % TH_N);
  const float* p  = pooled + (long)b * DM;
  const float* th = theme + (long)t * DM;
  float s = 0.0f;
  for (int d = 0; d < DM; ++d) s += p[d] * th[d];
  outp[i] = s;
}

} // namespace

// ---------------------------------------------------------------------------
extern "C" void kernel_launch(void* const* d_in, const int* in_sizes, int n_in,
                              void* d_out, int out_size, void* d_ws, size_t ws_size,
                              hipStream_t stream)
{
  (void)in_sizes; (void)n_in; (void)out_size; (void)ws_size;

  const int*   dec       = (const int*)  d_in[0];
  const float* mems      = (const float*)d_in[1];
  const float* rel_t_r   = (const float*)d_in[2];
  const float* rel_r_f   = (const float*)d_in[3];
  const float* theme_emb = (const float*)d_in[4];
  const float* role_emb  = (const float*)d_in[5];
  const float* form_emb  = (const float*)d_in[6];
  const float* word_emb  = (const float*)d_in[7];
  const float* theme_W   = (const float*)d_in[8];
  const float* theme_b   = (const float*)d_in[9];
  const float* r_w_bias  = (const float*)d_in[10];
  const float* r_r_bias  = (const float*)d_in[11];
  const float* qkv_W     = (const float*)d_in[12];
  const float* qkv_b     = (const float*)d_in[13];
  const float* r_W       = (const float*)d_in[14];
  const float* o_W       = (const float*)d_in[15];
  const float* o_b       = (const float*)d_in[16];
  const float* ln_att_g  = (const float*)d_in[17];
  const float* ln_att_b  = (const float*)d_in[18];
  const float* ln1_g     = (const float*)d_in[19];
  const float* ln1_b     = (const float*)d_in[20];
  const float* ffn_W1    = (const float*)d_in[21];
  const float* ffn_b1    = (const float*)d_in[22];
  const float* ffn_W2    = (const float*)d_in[23];
  const float* ffn_b2    = (const float*)d_in[24];
  const float* ln2_g     = (const float*)d_in[25];
  const float* ln2_b     = (const float*)d_in[26];

  char* base = (char*)d_ws;
  size_t off = 0;
  auto alloc = [&](size_t bytes) -> void* {
    void* p = base + off;
    off += (bytes + 255) & ~(size_t)255;
    return p;
  };

  // transposed bf16 weights (Bt layouts: [N][K])
  bf16*  qkvWt   = (bf16*) alloc((long)L_ * H3 * DM * 2);
  bf16*  rWt     = (bf16*) alloc((long)L_ * DM * DM * 2);
  bf16*  oWt     = (bf16*) alloc((long)L_ * DM * DM * 2);
  bf16*  W1t     = (bf16*) alloc((long)L_ * DI * DM * 2);
  bf16*  W2t     = (bf16*) alloc((long)L_ * DM * DI * 2);
  bf16*  themeWt = (bf16*) alloc((long)DM * 512 * 2);
  bf16*  pos_bf  = (bf16*) alloc((long)KL * DM * 2);
  bf16*  cat_bf  = (bf16*) alloc((long)KL * BS * DM * 2);
  float* heads   = (float*)alloc((long)KL * BS * H3 * 4);
  float* rh      = (float*)alloc((long)KL * DM * 4);
  bf16*  r_bf    = (bf16*) alloc((long)NHD * KL * DH * 2);
  bf16*  qw_bf   = (bf16*) alloc((long)BS * NHD * QL * DH * 2);
  bf16*  qr_bf   = (bf16*) alloc((long)BS * NHD * QL * DH * 2);
  bf16*  k_bf    = (bf16*) alloc((long)BS * NHD * KL * DH * 2);
  bf16*  vT_bf   = (bf16*) alloc((long)BS * NHD * DH * KL * 2);
  float* S1      = (float*)alloc((long)NHD * QL * KL * 4);
  float* S2      = (float*)alloc((long)NHD * QL * KL * 4);
  bf16*  P_bf    = (bf16*) alloc((long)NHD * QL * KL * 2);
  float* vecf    = (float*)alloc((long)QL * BS * DM * 4);
  bf16*  vec_bf  = (bf16*) alloc((long)QL * BS * DM * 2);
  float* att     = (float*)alloc((long)QL * BS * DM * 4);
  float* hbuf    = (float*)alloc((long)QL * BS * DM * 4);
  bf16*  a_bf    = (bf16*) alloc((long)QL * BS * DM * 2);
  bf16*  mid_bf  = (bf16*) alloc((long)QL * BS * DI * 2);
  float* tbuf    = (float*)alloc((long)QL * BS * DM * 4);
  float* outb    = (float*)alloc((long)QL * BS * DM * 4);
  float* role_cat= (float*)alloc((long)RO_N * 256 * 4);
  bf16*  tc_bf   = (bf16*) alloc((long)TH_N * 512 * 2);
  float* themef  = (float*)alloc((long)TH_N * DM * 4);
  float* pooled  = (float*)alloc((long)BS * DM * 4);

  // ---- weight fp32 -> bf16 transpose-conversions (recomputed each call)
  auto convT = [&](const float* src, bf16* dst, int K, int N, int layers) {
    k_f2bT<<<dim3(N / 32, K / 32, layers), 256, 0, stream>>>(
        src, dst, K, N, (long)K * N, (long)K * N);
  };
  convT(qkv_W,   qkvWt,   DM, H3, L_);
  convT(r_W,     rWt,     DM, DM, L_);
  convT(o_W,     oWt,     DM, DM, L_);
  convT(ffn_W1,  W1t,     DM, DI, L_);
  convT(ffn_W2,  W2t,     DI, DM, L_);
  convT(theme_W, themeWt, 512, DM, 1);

  // ---- theme relation pipeline
  k_role<<<(RO_N * 256) / 256, 256, 0, stream>>>(rel_r_f, form_emb, role_emb, role_cat);
  k_theme_cat<<<((long)TH_N * 512 + 255) / 256, 256, 0, stream>>>(rel_t_r, role_cat, theme_emb, tc_bf);
  gemm_k<128, 128, 0, 2><<<dim3(DM / 128, (TH_N + 127) / 128, 1), 256, 0, stream>>>(
      tc_bf, themeWt, themef, theme_b, TH_N, DM, 512, 512, 512, DM, 0, 0, 0);

  // ---- embeddings & positions
  k_embed<<<((long)QL * BS * DM) / 256, 256, 0, stream>>>(dec, word_emb, outb);
  k_pos<<<(KL * 512) / 256, 256, 0, stream>>>(pos_bf);

  // ---- layers
  for (int i = 0; i < L_; ++i) {
    k_cat<<<((long)KL * BS * DM) / 256, 256, 0, stream>>>(
        mems + (long)i * ML * BS * DM, outb, cat_bf);

    // heads = cat @ qkv_W[i] + qkv_b[i]   [16384 x 3072]
    gemm_k<128, 128, 0, 0><<<dim3(H3 / 128, (KL * BS) / 128, 1), 256, 0, stream>>>(
        cat_bf, qkvWt + (long)i * H3 * DM, heads, qkv_b + (long)i * H3,
        KL * BS, H3, DM, DM, DM, H3, 0, 0, 0);

    // rh = pos @ r_W[i]   [1024 x 1024]
    gemm_k<128, 128, 0, 0><<<dim3(DM / 128, KL / 128, 1), 256, 0, stream>>>(
        pos_bf, rWt + (long)i * DM * DM, rh, nullptr,
        KL, DM, DM, DM, DM, DM, 0, 0, 0);

    k_prep_r<<<((long)NHD * KL * DH) / 256, 256, 0, stream>>>(rh, r_bf);
    k_prep_qk<<<((long)BS * NHD * KL * DH) / 256, 256, 0, stream>>>(
        heads, r_w_bias, r_r_bias, qw_bf, qr_bf, k_bf);
    k_trans_v<<<dim3(KL / 32, DH / 32, BS * NHD), 256, 0, stream>>>(heads, vT_bf);

    for (int b = 0; b < BS; ++b) {
      // AC[n] = (q + r_w_bias) @ K^T      [16 x 512 x 1024]
      gemm_k<128, 128, 0, 0><<<dim3(KL / 128, QL / 128, NHD), 256, 0, stream>>>(
          qw_bf + (long)b * NHD * QL * DH, k_bf + (long)b * NHD * KL * DH, S1, nullptr,
          QL, KL, DH, DH, DH, KL, (long)QL * DH, (long)KL * DH, (long)QL * KL);
      // BD_raw[n] = (q + r_r_bias) @ R^T  [16 x 512 x 1024]
      gemm_k<128, 128, 0, 0><<<dim3(KL / 128, QL / 128, NHD), 256, 0, stream>>>(
          qr_bf + (long)b * NHD * QL * DH, r_bf, S2, nullptr,
          QL, KL, DH, DH, DH, KL, (long)QL * DH, (long)KL * DH, (long)QL * KL);
      // exact rel_shift + mask + softmax -> bf16 probs
      k_softmax<<<NHD * QL, 256, 0, stream>>>(S1, S2, dec, b, P_bf);
      // vec = P @ V  ->  [q][b][n*64+d] layout via ldc/strideC
      gemm_k<128, 64, 0, 0><<<dim3(1, QL / 128, NHD), 256, 0, stream>>>(
          P_bf, vT_bf + (long)b * NHD * DH * KL, vecf + (long)b * DM, nullptr,
          QL, DH, KL, KL, KL, BS * DM, (long)QL * KL, (long)DH * KL, (long)DH);
    }

    k_f2b<<<4096, 256, 0, stream>>>(vecf, vec_bf, (long)QL * BS * DM);

    // att = vec @ o_W[i] + o_b[i]
    gemm_k<128, 128, 0, 0><<<dim3(DM / 128, (QL * BS) / 128, 1), 256, 0, stream>>>(
        vec_bf, oWt + (long)i * DM * DM, att, o_b + (long)i * DM,
        QL * BS, DM, DM, DM, DM, DM, 0, 0, 0);

    // h = LN(out + att), a = LN(h) (bf16)
    k_add_ln2<<<QL * BS, 256, 0, stream>>>(outb, att,
        ln_att_g + (long)i * DM, ln_att_b + (long)i * DM,
        ln1_g + (long)i * DM, ln1_b + (long)i * DM, hbuf, a_bf);

    // mid = relu(a @ W1 + b1) (bf16)
    gemm_k<128, 128, 1, 1><<<dim3(DI / 128, (QL * BS) / 128, 1), 256, 0, stream>>>(
        a_bf, W1t + (long)i * DI * DM, mid_bf, ffn_b1 + (long)i * DI,
        QL * BS, DI, DM, DM, DM, DI, 0, 0, 0);

    // t = mid @ W2 + b2 (f32)
    gemm_k<128, 128, 0, 0><<<dim3(DM / 128, (QL * BS) / 128, 1), 256, 0, stream>>>(
        mid_bf, W2t + (long)i * DM * DI, tbuf, ffn_b2 + (long)i * DM,
        QL * BS, DM, DI, DI, DI, DM, 0, 0, 0);

    // out = LN(h + t)
    k_add_ln1<<<QL * BS, 256, 0, stream>>>(hbuf, tbuf,
        ln2_g + (long)i * DM, ln2_b + (long)i * DM, outb);
  }

  // ---- pool (tokens == E_INDEX) and theme similarity logits
  k_pool<<<(BS * DM) / 256, 256, 0, stream>>>(dec, outb, pooled);
  k_logits<<<((long)BS * TH_N + 255) / 256, 256, 0, stream>>>(pooled, themef, (float*)d_out);
}
